// Head_41798621724700
// MI455X (gfx1250) — matrix operand order repaired and checked by reading, not compiled, not run on hardware
//
#include <hip/hip_runtime.h>

typedef __attribute__((ext_vector_type(16))) _Float16 v16h;
typedef __attribute__((ext_vector_type(8)))  float    v8f;
typedef __attribute__((ext_vector_type(4)))  float    v4f;
typedef __attribute__((ext_vector_type(4)))  unsigned int v4u;
typedef __attribute__((ext_vector_type(8)))  int      v8i;
typedef __attribute__((ext_vector_type(4)))  int      v4i;

#define T_LEN   8192
#define CH      1024
#define HD      64
#define QT      512            // T/16 tiles
#define KSTEPS  32             // CH/32
#define NEG_INF (-1e30f)
#define SCALE   0.125f         // hd^-0.5

#define HAVE_TDM __has_builtin(__builtin_amdgcn_tensor_load_to_lds)

// ---- workspace layout (bytes) ----
#define WS_QA   0                                   // q A-frags: 512 tiles * 2 ksteps * 512 halves
#define WS_KB   (WS_QA + QT*1024*2)                 // k B-frags
#define WS_VB   (WS_KB + QT*1024*2)                 // v B-frags: 256 k32-tiles * 4 ntiles * 512 halves
#define WS_WF   (WS_VB + 256*4*512*2)               // W B-frags: 3 * 32 * 4 * 512 halves
#define WS_KF   (WS_WF + 3*32*4*512*2)              // k f32 row-major
#define WS_VF   (WS_KF + T_LEN*HD*4)                // v f32 row-major
#define WS_RM   (WS_VF + T_LEN*HD*4)                // row max
#define WS_RL   (WS_RM + T_LEN*4)                   // row sumexp
#define WS_CS   (WS_RL + T_LEN*4)                   // column sums
#define WS_EV   (WS_CS + T_LEN*4)                   // evict index (int)

static __device__ __forceinline__ v8f wmma_f16(v16h a, v16h b, v8f c) {
    return __builtin_amdgcn_wmma_f32_16x16x32_f16(false, a, false, b, (short)0, c, false, false);
}

// ---------------- kernel 0: repack weights into B-fragment layout ----------------
__global__ void wprep(const float* __restrict__ Wq, const float* __restrict__ Wk,
                      const float* __restrict__ Wv, _Float16* __restrict__ wf) {
    int idx = blockIdx.x * 256 + threadIdx.x;
    if (idx >= 3 * 32 * 4 * 512) return;
    int h    = idx & 15;
    int lane = (idx >> 4) & 31;
    int nt   = (idx >> 9) & 3;
    int kt   = (idx >> 11) & 31;
    int m    = idx >> 16;
    int k = kt * 32 + ((lane >= 16) ? 16 : 0) + h;   // B layout: lanes 0-15 -> K lo half
    int n = nt * 16 + (lane & 15);
    const float* W = (m == 0) ? Wq : ((m == 1) ? Wk : Wv);
    wf[idx] = (_Float16)W[k * HD + n];
}

// ---------------- kernel 1: q/k/v projection (WMMA) + RoPE + fragment packing ----
// The 16x1024 fp32 slab of x consumed by this block is DMA'd into LDS by the
// Tensor Data Mover (one tensor_load_to_lds, tracked by TENSORcnt), then the
// WMMA A-fragments are read from LDS while W B-fragments stream from global.
__global__ void __launch_bounds__(32) qkv_rope(const float* __restrict__ x,
                                               const _Float16* __restrict__ wf,
                                               _Float16* __restrict__ qA,
                                               _Float16* __restrict__ kB,
                                               _Float16* __restrict__ vB,
                                               float* __restrict__ kf,
                                               float* __restrict__ vf) {
    const int blk  = blockIdx.x;            // 16-row tile
    const int lane = threadIdx.x;
    const int t0   = blk * 16;
    const int row  = lane & 15;
    const bool hi  = lane >= 16;

    __shared__ float ldsx[16 * CH];         // 64 KB: x tile, later reused for repack

#if HAVE_TDM
    {
        // ---- Tensor DMA descriptor (D#), 2D tile: 1024 x 16 elements of 4 bytes ----
        unsigned lds_off = (unsigned)(uintptr_t)(void*)ldsx;  // LDS byte offset (low 32 bits)
        unsigned long long ga = (unsigned long long)(uintptr_t)(x + (size_t)t0 * CH);
        v4u g0;
        g0[0] = 1u;                                          // count=1, user descriptor
        g0[1] = lds_off;                                     // lds_addr
        g0[2] = (unsigned)(ga & 0xffffffffu);                // global_addr[31:0]
        g0[3] = (unsigned)((ga >> 32) & 0x1ffffffu) | 0x80000000u;  // global_addr[56:32] | type=2
        v8i g1;
        g1[0] = 2 << 16;                                     // data_size = 4 bytes
        g1[1] = (int)((CH & 0xffff) << 16);                  // tensor_dim0[15:0]
        g1[2] = (int)((CH >> 16) | (16 << 16));              // tensor_dim0[31:16] | tensor_dim1[15:0]=16
        g1[3] = (int)(CH << 16);                             // tensor_dim1[31:16]=0 | tile_dim0=1024
        g1[4] = 16;                                          // tile_dim1=16, tile_dim2=0
        g1[5] = CH;                                          // tensor_dim0_stride[31:0] = 1024
        g1[6] = 0;                                           // stride0 hi | stride1 lo
        g1[7] = 0;
        v4i g2 = {0, 0, 0, 0};                               // 2D tensor: groups 2/3 unused
        v4i g3 = {0, 0, 0, 0};
        v8i g4 = {0, 0, 0, 0, 0, 0, 0, 0};                   // extra group (6-arg toolchain form)
        __builtin_amdgcn_tensor_load_to_lds(g0, g1, g2, g3, g4, 0);
        __builtin_amdgcn_s_wait_tensorcnt(0);
    }
#else
    {
        const v4f* src = (const v4f*)(x + (size_t)t0 * CH);
        v4f* dst = (v4f*)ldsx;
        for (int i = lane; i < 16 * CH / 4; i += 32) dst[i] = src[i];
        __syncthreads();
    }
#endif

    v8f acc[3][4];
    for (int m = 0; m < 3; m++)
        for (int nt = 0; nt < 4; nt++)
            acc[m][nt] = (v8f){0.f, 0.f, 0.f, 0.f, 0.f, 0.f, 0.f, 0.f};

    for (int ks = 0; ks < KSTEPS; ks++) {
        const int k0 = ks * 32;
        // A-frag from LDS x tile (f32 -> f16), contiguous k order: k = 16*(lane/16) + s
        const int b1 = k0 + (hi ? 16 : 0);
        const int b2 = b1 + 8;
        const float* xr = ldsx + row * CH;
        v4f f0 = *(const v4f*)(xr + b1);
        v4f f1 = *(const v4f*)(xr + b1 + 4);
        v4f f2 = *(const v4f*)(xr + b2);
        v4f f3 = *(const v4f*)(xr + b2 + 4);
        v16h a;
        for (int i = 0; i < 4; i++) {
            a[i]      = (_Float16)f0[i];
            a[4 + i]  = (_Float16)f1[i];
            a[8 + i]  = (_Float16)f2[i];
            a[12 + i] = (_Float16)f3[i];
        }
        for (int m = 0; m < 3; m++)
            for (int nt = 0; nt < 4; nt++) {
                v16h b = *(const v16h*)(wf + ((size_t)((m * 32 + ks) * 4 + nt)) * 512 + lane * 16);
                acc[m][nt] = wmma_f16(a, b, acc[m][nt]);
            }
    }

    // RoPE on q (acc[0]) and k (acc[1]); adjacent columns live in adjacent lanes
    for (int mm = 0; mm < 2; mm++) {
        for (int nt = 0; nt < 4; nt++)
            for (int r = 0; r < 8; r++) {
                float v  = acc[mm][nt][r];
                float pv = __shfl_xor(v, 1);
                int col  = nt * 16 + (lane & 15);
                int p    = col >> 1;
                int t    = t0 + r + (hi ? 8 : 0);
                float ang = (float)t * __powf(10000.0f, -(float)p / 32.0f);
                float cs = __cosf(ang), sn = __sinf(ang);
                acc[mm][nt][r] = ((lane & 1) == 0) ? (v * cs - pv * sn)
                                                   : (pv * sn + v * cs);
            }
    }
    for (int nt = 0; nt < 4; nt++) acc[0][nt] *= SCALE;  // fold score scale into q

    // f32 row-major copies of k and v (for cache gather)
    for (int nt = 0; nt < 4; nt++)
        for (int r = 0; r < 8; r++) {
            int rr  = r + (hi ? 8 : 0);
            int col = nt * 16 + (lane & 15);
            kf[(size_t)(t0 + rr) * HD + col] = acc[1][nt][r];
            vf[(size_t)(t0 + rr) * HD + col] = acc[2][nt][r];
        }

    float* lds = ldsx;  // x tile no longer needed; reuse LDS for fragment repack
    __syncthreads();
    // ---- q -> A-fragment layout ----
    for (int nt = 0; nt < 4; nt++)
        for (int r = 0; r < 8; r++)
            lds[(r + (hi ? 8 : 0)) * 64 + nt * 16 + (lane & 15)] = acc[0][nt][r];
    __syncthreads();
    for (int i = lane; i < 1024; i += 32) {
        int r = i >> 6, c = i & 63;
        int kstep = c >> 5, kk = c & 31;
        int dl, h;
        if (kk < 8)       { dl = r;      h = kk; }
        else if (kk < 16) { dl = r + 16; h = kk - 8; }
        else if (kk < 24) { dl = r;      h = kk - 8; }
        else              { dl = r + 16; h = kk - 16; }
        qA[(size_t)(blk * 2 + kstep) * 512 + dl * 16 + h] = (_Float16)lds[i];
    }
    __syncthreads();
    // ---- k -> B-fragment layout ----
    for (int nt = 0; nt < 4; nt++)
        for (int r = 0; r < 8; r++)
            lds[(r + (hi ? 8 : 0)) * 64 + nt * 16 + (lane & 15)] = acc[1][nt][r];
    __syncthreads();
    for (int i = lane; i < 1024; i += 32) {
        int r = i >> 6, c = i & 63;
        int kstep = c >> 5, kk = c & 31;
        int dl = r + (((kk >> 3) & 1) ? 16 : 0);
        int h  = (kk & 7) + ((kk >= 16) ? 8 : 0);
        kB[(size_t)(blk * 2 + kstep) * 512 + dl * 16 + h] = (_Float16)lds[i];
    }
    __syncthreads();
    // ---- v -> B-fragment layout (K dim = key index, 32-key blocks) ----
    for (int nt = 0; nt < 4; nt++)
        for (int r = 0; r < 8; r++)
            lds[(r + (hi ? 8 : 0)) * 64 + nt * 16 + (lane & 15)] = acc[2][nt][r];
    __syncthreads();
    {
        int k32  = blk >> 1;
        int half = blk & 1;
        for (int i = lane; i < 1024; i += 32) {
            int r = i >> 6, c = i & 63;
            int nt = c >> 4, n = c & 15;
            int dl = n + (half ? 16 : 0);
            vB[(size_t)(k32 * 4 + nt) * 512 + dl * 16 + r] = (_Float16)lds[i];
        }
    }
}

// ---------------- kernel 2: flash-style softmax stats per query row ----------------
__global__ void __launch_bounds__(32) attn_stats(const _Float16* __restrict__ qA,
                                                 const _Float16* __restrict__ kB,
                                                 float* __restrict__ rowm,
                                                 float* __restrict__ rowl) {
    const int qt = blockIdx.x, lane = threadIdx.x;
    const bool hi = lane >= 16;
    const int col = lane & 15;
    v16h a0 = *(const v16h*)(qA + (size_t)(qt * 2 + 0) * 512 + lane * 16);
    v16h a1 = *(const v16h*)(qA + (size_t)(qt * 2 + 1) * 512 + lane * 16);
    float m_run[8], l_run[8];
    for (int r = 0; r < 8; r++) { m_run[r] = -3.0e38f; l_run[r] = 0.f; }
    for (int kt = 0; kt <= qt; kt++) {
        if (kt + 8 <= qt)  // warm L2/WGP$ for the k-fragment stream
            __builtin_prefetch(kB + (size_t)((kt + 8) * 2) * 512 + lane * 16, 0, 0);
        v16h b0 = *(const v16h*)(kB + (size_t)(kt * 2 + 0) * 512 + lane * 16);
        v16h b1 = *(const v16h*)(kB + (size_t)(kt * 2 + 1) * 512 + lane * 16);
        v8f c = (v8f){0.f, 0.f, 0.f, 0.f, 0.f, 0.f, 0.f, 0.f};
        c = wmma_f16(a0, b0, c);
        c = wmma_f16(a1, b1, c);
        if (kt == qt)
            for (int r = 0; r < 8; r++)
                if (col > r + (hi ? 8 : 0)) c[r] = NEG_INF;
        for (int r = 0; r < 8; r++) {
            float tv = c[r];
            float tmax = tv;
            for (int s = 1; s < 16; s <<= 1) tmax = fmaxf(tmax, __shfl_xor(tmax, s));
            float m_new = fmaxf(m_run[r], tmax);
            float p = __expf(tv - m_new);
            for (int s = 1; s < 16; s <<= 1) p += __shfl_xor(p, s);
            l_run[r] = l_run[r] * __expf(m_run[r] - m_new) + p;
            m_run[r] = m_new;
        }
    }
    if (col == 0) {
        int off = hi ? 8 : 0;
        for (int r = 0; r < 8; r++) {
            rowm[qt * 16 + off + r] = m_run[r];
            rowl[qt * 16 + off + r] = l_run[r];
        }
    }
}

// ---------------- kernel 3: column sums of wei ----------------
__global__ void __launch_bounds__(32) col_sums(const _Float16* __restrict__ qA,
                                               const _Float16* __restrict__ kB,
                                               const float* __restrict__ rowm,
                                               const float* __restrict__ rowl,
                                               float* __restrict__ cs) {
    const int kt = blockIdx.x, lane = threadIdx.x;
    const bool hi = lane >= 16;
    const int col = lane & 15;
    const int off = hi ? 8 : 0;
    v16h b0 = *(const v16h*)(kB + (size_t)(kt * 2 + 0) * 512 + lane * 16);
    v16h b1 = *(const v16h*)(kB + (size_t)(kt * 2 + 1) * 512 + lane * 16);
    float accum = 0.f;
    for (int qt = kt; qt < QT; qt++) {
        v16h a0 = *(const v16h*)(qA + (size_t)(qt * 2 + 0) * 512 + lane * 16);
        v16h a1 = *(const v16h*)(qA + (size_t)(qt * 2 + 1) * 512 + lane * 16);
        v8f c = (v8f){0.f, 0.f, 0.f, 0.f, 0.f, 0.f, 0.f, 0.f};
        c = wmma_f16(a0, b0, c);
        c = wmma_f16(a1, b1, c);
        if (qt == kt)
            for (int r = 0; r < 8; r++)
                if (col > r + off) c[r] = NEG_INF;
        for (int r = 0; r < 8; r++) {
            float m = rowm[qt * 16 + off + r];
            float l = rowl[qt * 16 + off + r];
            accum += __expf(c[r] - m) / l;
        }
    }
    accum += __shfl_xor(accum, 16);
    if (!hi) cs[kt * 16 + col] = accum;
}

// ---------------- kernel 4: argmin with first-index tie break ----------------
__global__ void __launch_bounds__(256) argmin_k(const float* __restrict__ cs, int* __restrict__ evict) {
    __shared__ float sv[256];
    __shared__ int   si[256];
    int tid = threadIdx.x;
    float best = 3.0e38f;
    int bi = 0;
    for (int i = tid; i < T_LEN; i += 256) {
        float v = cs[i];
        if (v < best) { best = v; bi = i; }
    }
    sv[tid] = best; si[tid] = bi;
    __syncthreads();
    for (int s = 128; s > 0; s >>= 1) {
        if (tid < s) {
            if (sv[tid + s] < sv[tid] || (sv[tid + s] == sv[tid] && si[tid + s] < si[tid])) {
                sv[tid] = sv[tid + s];
                si[tid] = si[tid + s];
            }
        }
        __syncthreads();
    }
    if (tid == 0) *evict = si[0];
}

// ---------------- kernel 5: wei -> importance (row-shifted) + out = wei@v ----------
__global__ void __launch_bounds__(32) attn_out(const _Float16* __restrict__ qA,
                                               const _Float16* __restrict__ kB,
                                               const _Float16* __restrict__ vB,
                                               const float* __restrict__ rowm,
                                               const float* __restrict__ rowl,
                                               const int* __restrict__ evictp,
                                               float* __restrict__ outp,
                                               float* __restrict__ imp) {
    const int qt = blockIdx.x, lane = threadIdx.x;
    const bool hi = lane >= 16;
    const int col = lane & 15;
    const int off = hi ? 8 : 0;
    const int evict = *evictp;
    v16h a0 = *(const v16h*)(qA + (size_t)(qt * 2 + 0) * 512 + lane * 16);
    v16h a1 = *(const v16h*)(qA + (size_t)(qt * 2 + 1) * 512 + lane * 16);
    float rm[8], rl[8];
    for (int r = 0; r < 8; r++) {
        rm[r] = rowm[qt * 16 + off + r];
        rl[r] = rowl[qt * 16 + off + r];
    }
    v8f outacc[4];
    for (int nt = 0; nt < 4; nt++)
        outacc[nt] = (v8f){0.f, 0.f, 0.f, 0.f, 0.f, 0.f, 0.f, 0.f};

    __shared__ _Float16 ldsw[16 * 32];
    const int nk32 = qt / 2 + 1;
    for (int k32 = 0; k32 < nk32; k32++) {
        for (int sub = 0; sub < 2; sub++) {
            int kt = k32 * 2 + sub;
            if (kt <= qt) {
                v16h b0 = *(const v16h*)(kB + (size_t)(kt * 2 + 0) * 512 + lane * 16);
                v16h b1 = *(const v16h*)(kB + (size_t)(kt * 2 + 1) * 512 + lane * 16);
                v8f c = (v8f){0.f, 0.f, 0.f, 0.f, 0.f, 0.f, 0.f, 0.f};
                c = wmma_f16(a0, b0, c);
                c = wmma_f16(a1, b1, c);
                if (kt == qt)
                    for (int r = 0; r < 8; r++)
                        if (col > r + off) c[r] = NEG_INF;
                for (int r = 0; r < 8; r++) {
                    float w = __expf(c[r] - rm[r]) / rl[r];
                    ldsw[(r + off) * 32 + sub * 16 + col] = (_Float16)w;
                    int qrow = qt * 16 + off + r;
                    if (qrow != evict) {
                        int drow = qrow - ((qrow > evict) ? 1 : 0);
                        imp[(size_t)drow * T_LEN + kt * 16 + col] = w;
                    }
                }
            } else {
                for (int r = 0; r < 8; r++)
                    ldsw[(r + off) * 32 + sub * 16 + col] = (_Float16)0.f;
            }
        }
        __syncthreads();
        // re-read wei as a 16x32 A-fragment
        v16h aw;
        int m = lane & 15;
        for (int h = 0; h < 16; h++) {
            int kk = (hi ? 16 : 0) + h;
            aw[h] = ldsw[m * 32 + kk];
        }
        for (int nt = 0; nt < 4; nt++) {
            v16h bv = *(const v16h*)(vB + (size_t)(k32 * 4 + nt) * 512 + lane * 16);
            outacc[nt] = wmma_f16(aw, bv, outacc[nt]);
        }
        __syncthreads();
    }
    for (int nt = 0; nt < 4; nt++)
        for (int r = 0; r < 8; r++)
            outp[(size_t)(qt * 16 + off + r) * HD + nt * 16 + col] = outacc[nt][r];
}

// ---------------- kernel 6: gather k/v cache minus evicted row ----------------
__global__ void gather_cache(const float* __restrict__ kf, const float* __restrict__ vf,
                             const int* __restrict__ evictp,
                             float* __restrict__ ck, float* __restrict__ cv) {
    int i = blockIdx.x * 256 + threadIdx.x;
    const int n = (T_LEN - 1) * HD;
    if (i >= 2 * n) return;
    int evict = *evictp;
    if (i < n) {
        int t = i / HD, c = i % HD;
        int s = t + (t >= evict ? 1 : 0);
        ck[i] = kf[(size_t)s * HD + c];
    } else {
        int j = i - n;
        int t = j / HD, c = j % HD;
        int s = t + (t >= evict ? 1 : 0);
        cv[j] = vf[(size_t)s * HD + c];
    }
}

extern "C" void kernel_launch(void* const* d_in, const int* in_sizes, int n_in,
                              void* d_out, int out_size, void* d_ws, size_t ws_size,
                              hipStream_t stream) {
    const float* x  = (const float*)d_in[0];
    const float* Wk = (const float*)d_in[1];
    const float* Wq = (const float*)d_in[2];
    const float* Wv = (const float*)d_in[3];

    char* ws = (char*)d_ws;
    _Float16* qA = (_Float16*)(ws + WS_QA);
    _Float16* kB = (_Float16*)(ws + WS_KB);
    _Float16* vB = (_Float16*)(ws + WS_VB);
    _Float16* wf = (_Float16*)(ws + WS_WF);
    float* kf   = (float*)(ws + WS_KF);
    float* vf   = (float*)(ws + WS_VF);
    float* rowm = (float*)(ws + WS_RM);
    float* rowl = (float*)(ws + WS_RL);
    float* cs   = (float*)(ws + WS_CS);
    int*   ev   = (int*)(ws + WS_EV);

    float* outp = (float*)d_out;
    float* imp  = outp + (size_t)T_LEN * HD;
    float* ck   = imp + (size_t)(T_LEN - 1) * T_LEN;
    float* cv   = ck + (size_t)(T_LEN - 1) * HD;

    // zero the above-diagonal part of importance (softmax zeros there)
    (void)hipMemsetAsync(imp, 0, (size_t)(T_LEN - 1) * T_LEN * sizeof(float), stream);

    wprep<<<(3 * 32 * 4 * 512 + 255) / 256, 256, 0, stream>>>(Wq, Wk, Wv, wf);
    qkv_rope<<<QT, 32, 0, stream>>>(x, wf, qA, kB, vB, kf, vf);
    attn_stats<<<QT, 32, 0, stream>>>(qA, kB, rowm, rowl);
    col_sums<<<QT, 32, 0, stream>>>(qA, kB, rowm, rowl, cs);
    argmin_k<<<1, 256, 0, stream>>>(cs, ev);
    attn_out<<<QT, 32, 0, stream>>>(qA, kB, vB, rowm, rowl, ev, outp, imp);
    gather_cache<<<(2 * (T_LEN - 1) * HD + 255) / 256, 256, 0, stream>>>(kf, vf, ev, ck, cv);
}
